// MultiHeadAttention_64716567216823
// MI455X (gfx1250) — matrix operand order repaired and checked by reading, not compiled, not run on hardware
//
#include <hip/hip_runtime.h>
#include <math.h>

#define D_MODEL 1024
#define N_HEADS 16
#define D_K     64
#define BATCH   4
#define SEQ     2048
#define TOKENS  (BATCH * SEQ)   // 8192

typedef __attribute__((ext_vector_type(16))) __bf16 v16bf;
typedef __attribute__((ext_vector_type(8)))  float  v8f;
typedef __attribute__((ext_vector_type(4)))  unsigned int u32x4;
typedef __attribute__((ext_vector_type(8)))  int i32x8;
typedef __attribute__((ext_vector_type(4)))  int i32x4;

#if __has_builtin(__builtin_amdgcn_tensor_load_to_lds)
#define USE_TDM 1
#else
#define USE_TDM 0
#endif

union FragB16 { v16bf v; unsigned int u[8]; unsigned short s[16]; };

static __device__ inline unsigned short f32_to_bf16_rne(float x) {
  union { float f; unsigned int u; } c; c.f = x;
  unsigned int lsb = (c.u >> 16) & 1u;
  unsigned int r = c.u + 0x7FFFu + lsb;
  return (unsigned short)(r >> 16);
}

static __device__ inline v8f zero8() {
  v8f z;
  #pragma unroll
  for (int i = 0; i < 8; ++i) z[i] = 0.0f;
  return z;
}

#if USE_TDM
// Issue one TDM 2D tile load (bf16 elements) into LDS.
// D# per CDNA5 ISA ch.8: group0 {count|lds_addr|global_addr|type=2},
// group1 {data_size=2B, tensor dims/strides (element units), tile dims}.
// clang-23 builtin form: (u32x4 g0, i32x8 g1, i32x4 g2, i32x4 g3, i32x8, i32 cpol)
static __device__ inline void tdm_load_tile_2d(unsigned long long gaddr,
                                               unsigned lds_off,
                                               unsigned tensor_w,   // dim0 (elems)
                                               unsigned tensor_h,   // dim1 (rows)
                                               unsigned stride_w,   // dim0 stride
                                               unsigned tile_w, unsigned tile_h) {
  u32x4 g0;
  g0[0] = 1u;                                   // count=1 (valid), user mode
  g0[1] = lds_off;                              // LDS byte address
  g0[2] = (unsigned)(gaddr & 0xFFFFFFFFull);    // global_addr[31:0]
  g0[3] = (unsigned)((gaddr >> 32) & 0x1FFFFFFull) | (2u << 30);  // [56:32]|type=2
  i32x8 g1;
  g1[0] = (int)(1u << 16);                                        // data_size=2B
  g1[1] = (int)((tensor_w & 0xFFFFu) << 16);                      // dim0[15:0]
  g1[2] = (int)(((tensor_w >> 16) & 0xFFFFu) | ((tensor_h & 0xFFFFu) << 16));
  g1[3] = (int)(((tensor_h >> 16) & 0xFFFFu) | (tile_w << 16));   // tile_dim0
  g1[4] = (int)(tile_h & 0xFFFFu);                                // tile_dim1
  g1[5] = (int)stride_w;                                          // dim0_stride lo
  g1[6] = 0;                                                      // stride hi / dim1_stride
  g1[7] = 0;
  i32x4 z4 = {0, 0, 0, 0};
  i32x8 z8 = {0, 0, 0, 0, 0, 0, 0, 0};
  __builtin_amdgcn_tensor_load_to_lds(g0, g1, z4, z4, z8, 0);
}
#endif

// ---------------------------------------------------------------------------
// fp32 -> bf16 (row-major, for x)
// ---------------------------------------------------------------------------
__global__ void cvt_f32_bf16(const float* __restrict__ in,
                             unsigned short* __restrict__ out, int n) {
  int i = blockIdx.x * blockDim.x + threadIdx.x;
  if (i < n) out[i] = f32_to_bf16_rne(in[i]);
}

// ---------------------------------------------------------------------------
// fp32 [K,N] -> bf16 transposed [N,K] (weights), LDS-tiled 32x32
// ---------------------------------------------------------------------------
__global__ __launch_bounds__(256)
void cvt_transpose_bf16(const float* __restrict__ in,
                        unsigned short* __restrict__ out, int K, int N) {
  __shared__ unsigned short tile[32][33];
  int n0 = blockIdx.x * 32;
  int k0 = blockIdx.y * 32;
  int tx = threadIdx.x & 31;
  int ty = threadIdx.x >> 5;  // 0..7
  #pragma unroll
  for (int j = 0; j < 4; ++j) {
    int kk = ty + j * 8;
    tile[kk][tx] = f32_to_bf16_rne(in[(size_t)(k0 + kk) * N + n0 + tx]);
  }
  __syncthreads();
  #pragma unroll
  for (int j = 0; j < 4; ++j) {
    int nn = ty + j * 8;
    out[(size_t)(n0 + nn) * K + k0 + tx] = tile[tx][nn];
  }
}

// ---------------------------------------------------------------------------
// bf16 GEMM: C[M,N] = A[M,K] @ Wt[N,K]^T + bias[N]
//   A row-major [M,K] bf16; Wt row-major [N,K] bf16 (pre-transposed weight)
// MODE 0: bf16 out, head-major [B,H,S,Dk]   (q, k)
// MODE 1: f32  out, row-major  [M,N]        (final projection)
// MODE 2: bf16 out, d-major    [B,H,Dk,S]   (v, transposed for flash PV)
// Block: 256 thr = 8 waves over 128x128 tile; wave = 64(M) x 32(N).
// LDS tiles filled by the Tensor Data Mover (one wave issues 2 descriptors,
// waits TENSORcnt==0, barrier releases the WMMA waves).
// ---------------------------------------------------------------------------
#define MT 128
#define NT 128
#define KT 32

template <int MODE>
__global__ __launch_bounds__(256)
void gemm_bf16(const unsigned short* __restrict__ A,
               const unsigned short* __restrict__ Wt,
               const float* __restrict__ bias,
               void* __restrict__ out, int M, int N, int K) {
  __shared__ unsigned int ldsA[MT * (KT / 2)];   // [128][16] u32 (bf16 pairs)
  __shared__ unsigned int ldsBt[NT * (KT / 2)];  // [128][16] u32

  const int tid  = threadIdx.x;
  const int lane = tid & 31;
  const int wave = tid >> 5;   // 0..7
  const int wm   = wave & 1;   // 2 wave rows (64 each)
  const int wn   = wave >> 1;  // 4 wave cols (32 each)
  const int m0   = blockIdx.x * MT;
  const int n0   = blockIdx.y * NT;
  const int half = lane >> 4;
  const int l15  = lane & 15;

  v8f acc[4][2];
  #pragma unroll
  for (int a = 0; a < 4; ++a)
    #pragma unroll
    for (int b = 0; b < 2; ++b) acc[a][b] = zero8();

  for (int k0 = 0; k0 < K; k0 += KT) {
#if USE_TDM
    if (tid < 32) {  // wave 0 drives the TDM; EXEC-independent DMA issue
      unsigned long long gaA =
          (unsigned long long)(uintptr_t)A + ((size_t)m0 * K + k0) * 2u;
      unsigned long long gaB =
          (unsigned long long)(uintptr_t)Wt + ((size_t)n0 * K + k0) * 2u;
      tdm_load_tile_2d(gaA, (unsigned)(size_t)ldsA,  K, M, K, KT, MT);
      tdm_load_tile_2d(gaB, (unsigned)(size_t)ldsBt, K, N, K, KT, NT);
      __builtin_amdgcn_s_wait_tensorcnt(0);
    }
#else
    {
      const unsigned int* Au = (const unsigned int*)A;
      const unsigned int* Wu = (const unsigned int*)Wt;
      int row = tid >> 1;
      int cb  = (tid & 1) * 8;
      const unsigned int* srcA = Au + (((size_t)(m0 + row) * K + k0) >> 1);
      const unsigned int* srcB = Wu + (((size_t)(n0 + row) * K + k0) >> 1);
      #pragma unroll
      for (int c = 0; c < 8; ++c) ldsA[row * 16 + cb + c] = srcA[cb + c];
      #pragma unroll
      for (int c = 0; c < 8; ++c) ldsBt[row * 16 + cb + c] = srcB[cb + c];
    }
#endif
    __syncthreads();

    // A frags: element e -> k = half*16 + e (contiguous 8 u32 per lane)
    FragB16 afr[4];
    #pragma unroll
    for (int am = 0; am < 4; ++am) {
      int r = wm * 64 + am * 16 + l15;
      #pragma unroll
      for (int p = 0; p < 8; ++p) {
        int ku = half * 8 + p;
        afr[am].u[p] = ldsA[r * 16 + ku];
      }
    }
    // B frags: element e -> k = half*16 + e (contiguous 8 u32 per lane)
    FragB16 bfr[2];
    #pragma unroll
    for (int bn = 0; bn < 2; ++bn) {
      int n = wn * 32 + bn * 16 + l15;
      #pragma unroll
      for (int p = 0; p < 8; ++p)
        bfr[bn].u[p] = ldsBt[n * 16 + half * 8 + p];
    }
    #pragma unroll
    for (int am = 0; am < 4; ++am)
      #pragma unroll
      for (int bn = 0; bn < 2; ++bn)
        acc[am][bn] = __builtin_amdgcn_wmma_f32_16x16x32_bf16(
            false, afr[am].v, false, bfr[bn].v, (short)0, acc[am][bn],
            false, false);
    __syncthreads();
  }

  // Epilogue: C element i -> row = i + 8*half, col = l15
  #pragma unroll
  for (int am = 0; am < 4; ++am) {
    #pragma unroll
    for (int bn = 0; bn < 2; ++bn) {
      int col = n0 + wn * 32 + bn * 16 + l15;
      float bv = bias[col];
      #pragma unroll
      for (int i = 0; i < 8; ++i) {
        int row = m0 + wm * 64 + am * 16 + half * 8 + i;
        float val = acc[am][bn][i] + bv;
        if (MODE == 0) {
          int b = row >> 11, s = row & (SEQ - 1);
          int h = col >> 6,  d = col & (D_K - 1);
          ((unsigned short*)out)[(((size_t)(b * N_HEADS + h) * SEQ + s) << 6) + d] =
              f32_to_bf16_rne(val);
        } else if (MODE == 1) {
          ((float*)out)[(size_t)row * N + col] = val;
        } else {  // MODE 2: [B,H,Dk,S]
          int b = row >> 11, s = row & (SEQ - 1);
          int h = col >> 6,  d = col & (D_K - 1);
          ((unsigned short*)out)[((size_t)((b * N_HEADS + h) * D_K + d) << 11) + s] =
              f32_to_bf16_rne(val);
        }
      }
    }
  }
}

// ---------------------------------------------------------------------------
// Flash attention: one wave per 16-row q-tile of one (b,h); 64-key steps.
// q,k: [B,H,S,64] bf16; v: [B,H,64,S] bf16 (d-major).
// Per step: 8 score WMMAs, online softmax, bf16 LDS relayout, 8 PV WMMAs.
// ---------------------------------------------------------------------------
__global__ __launch_bounds__(256)
void flash_attn(const unsigned short* __restrict__ q,
                const unsigned short* __restrict__ k,
                const unsigned short* __restrict__ v,
                unsigned short* __restrict__ ctx) {  // [B,S,1024] bf16
  __shared__ unsigned short pstage[8][16 * 64];  // per-wave 16x64 bf16 tile

  const int tid  = threadIdx.x;
  const int lane = tid & 31;
  const int wave = tid >> 5;
  const int half = lane >> 4;
  const int l15  = lane & 15;

  int gw = blockIdx.x * 8 + wave;
  int qt = gw & (SEQ / 16 - 1);  // 128 q-tiles per (b,h)
  int bh = gw >> 7;              // 0..63
  const int q0 = qt * 16;

  const unsigned short* kbase = k + (size_t)bh * SEQ * D_K;
  const unsigned short* vbase = v + (size_t)bh * D_K * SEQ;
  const unsigned int* qu  = (const unsigned int*)(q + (size_t)bh * SEQ * D_K);
  const unsigned int* ku  = (const unsigned int*)kbase;
  const unsigned int* vtu = (const unsigned int*)vbase;

  // Q fragments (A-layout), d 0..31 / 32..63
  FragB16 aq[2];
  {
    size_t r = (size_t)(q0 + l15);
    #pragma unroll
    for (int f = 0; f < 2; ++f)
      #pragma unroll
      for (int p = 0; p < 8; ++p) {
        int du = f * 16 + half * 8 + p;
        aq[f].u[p] = qu[r * 32 + du];
      }
  }

  v8f acc[4];
  #pragma unroll
  for (int f = 0; f < 4; ++f) acc[f] = zero8();
  float mrow[8], lrow[8];
  #pragma unroll
  for (int i = 0; i < 8; ++i) { mrow[i] = -3.0e38f; lrow[i] = 0.0f; }

  unsigned short* myp = pstage[wave];
  unsigned int*   mypu = (unsigned int*)myp;
  const float scale = 0.125f;  // 1/sqrt(64)

  for (int t0 = 0; t0 < SEQ; t0 += 64) {
    // Prefetch next key block (global_prefetch_b8; speculative, cacheline pulls)
    if (t0 + 64 < SEQ) {
      __builtin_prefetch(kbase + (size_t)(t0 + 64 + lane * 2) * D_K, 0, 1);
      __builtin_prefetch(vbase + (size_t)(lane * 2) * SEQ + t0 + 64, 0, 1);
    }

    // ---- scores: 4 key groups of 16, each = 2 chained WMMAs over d ----
    v8f s[4];
    #pragma unroll
    for (int g = 0; g < 4; ++g) {
      FragB16 bk0, bk1;  // B = K^T: element e -> d = half*16+e (contiguous)
      const unsigned int* krow = ku + (size_t)(t0 + g * 16 + l15) * 32;
      #pragma unroll
      for (int p = 0; p < 8; ++p) bk0.u[p] = krow[half * 8 + p];
      #pragma unroll
      for (int p = 0; p < 8; ++p) bk1.u[p] = krow[16 + half * 8 + p];
      v8f sg = zero8();
      sg = __builtin_amdgcn_wmma_f32_16x16x32_bf16(false, aq[0].v, false, bk0.v, (short)0, sg, false, false);
      sg = __builtin_amdgcn_wmma_f32_16x16x32_bf16(false, aq[1].v, false, bk1.v, (short)0, sg, false, false);
      s[g] = sg;
    }

    // ---- online softmax (row = i + 8*half; 16 lanes/half share a row) ----
    float alpha[8];
    #pragma unroll
    for (int i = 0; i < 8; ++i) {
      float a0 = s[0][i] * scale, a1 = s[1][i] * scale;
      float a2 = s[2][i] * scale, a3 = s[3][i] * scale;
      float mx = fmaxf(fmaxf(a0, a1), fmaxf(a2, a3));
      #pragma unroll
      for (int m = 1; m <= 8; m <<= 1) mx = fmaxf(mx, __shfl_xor(mx, m, 32));
      float mnew = fmaxf(mrow[i], mx);
      float scl  = __expf(mrow[i] - mnew);
      float p0 = __expf(a0 - mnew), p1 = __expf(a1 - mnew);
      float p2 = __expf(a2 - mnew), p3 = __expf(a3 - mnew);
      float ps = (p0 + p1) + (p2 + p3);
      #pragma unroll
      for (int m = 1; m <= 8; m <<= 1) ps += __shfl_xor(ps, m, 32);
      lrow[i]  = lrow[i] * scl + ps;
      mrow[i]  = mnew;
      alpha[i] = scl;
      s[0][i] = p0; s[1][i] = p1; s[2][i] = p2; s[3][i] = p3;
    }
    #pragma unroll
    for (int f = 0; f < 4; ++f)
      #pragma unroll
      for (int i = 0; i < 8; ++i) acc[f][i] *= alpha[i];

    // ---- relayout P: C-layout f32 -> bf16 LDS tile [16][64] -> A-frags ----
    #pragma unroll
    for (int g = 0; g < 4; ++g)
      #pragma unroll
      for (int i = 0; i < 8; ++i) {
        int row = half * 8 + i;
        myp[row * 64 + g * 16 + l15] = f32_to_bf16_rne(s[g][i]);
      }
    __builtin_amdgcn_wave_barrier();
    asm volatile("s_wait_dscnt 0x0" ::: "memory");
    FragB16 ap[2];  // P A-frags: keys 0..31 / 32..63 of this block
    #pragma unroll
    for (int j = 0; j < 2; ++j)
      #pragma unroll
      for (int p = 0; p < 8; ++p) {
        int ku32 = j * 16 + half * 8 + p;
        ap[j].u[p] = mypu[l15 * 32 + ku32];
      }
    __builtin_amdgcn_wave_barrier();

    // ---- P @ V with d-major V: per-lane contiguous key runs ----
    #pragma unroll
    for (int f = 0; f < 4; ++f) {
      #pragma unroll
      for (int j = 0; j < 2; ++j) {
        FragB16 bv;  // element e -> key = j*32 + half*16 + e, col d = f*16+l15
        const unsigned int* vrow =
            vtu + (size_t)(f * 16 + l15) * (SEQ / 2) + ((t0 + j * 32) >> 1);
        #pragma unroll
        for (int p = 0; p < 8; ++p) bv.u[p] = vrow[half * 8 + p];
        acc[f] = __builtin_amdgcn_wmma_f32_16x16x32_bf16(
            false, ap[j].v, false, bv.v, (short)0, acc[f], false, false);
      }
    }
  }

  // ---- normalize, scatter ctx [B,S,H*64] bf16 ----
  int b = bh >> 4, h = bh & (N_HEADS - 1);
  #pragma unroll
  for (int f = 0; f < 4; ++f) {
    #pragma unroll
    for (int i = 0; i < 8; ++i) {
      int row = q0 + half * 8 + i;
      float val = acc[f][i] / lrow[i];
      int d = f * 16 + l15;
      ctx[(size_t)(b * SEQ + row) * D_MODEL + h * D_K + d] = f32_to_bf16_rne(val);
    }
  }
}

// ---------------------------------------------------------------------------
extern "C" void kernel_launch(void* const* d_in, const int* in_sizes, int n_in,
                              void* d_out, int out_size, void* d_ws, size_t ws_size,
                              hipStream_t stream) {
  const float* x  = (const float*)d_in[0];
  const float* Wq = (const float*)d_in[1];
  const float* bq = (const float*)d_in[2];
  const float* Wk = (const float*)d_in[3];
  const float* bk = (const float*)d_in[4];
  const float* Wv = (const float*)d_in[5];
  const float* bv = (const float*)d_in[6];
  const float* Wo = (const float*)d_in[7];
  const float* bo = (const float*)d_in[8];

  const size_t NX = (size_t)TOKENS * D_MODEL;   // 8.39M elems
  const size_t NW = (size_t)D_MODEL * D_MODEL;  // 1.05M elems
  unsigned short* xb   = (unsigned short*)d_ws;
  unsigned short* Wqt  = xb  + NX;   // transposed bf16 weights [N,K]
  unsigned short* Wkt  = Wqt + NW;
  unsigned short* Wvt  = Wkt + NW;
  unsigned short* Wot  = Wvt + NW;
  unsigned short* qb   = Wot + NW;   // [B,H,S,Dk]
  unsigned short* kb   = qb  + NX;   // [B,H,S,Dk]
  unsigned short* vt   = kb  + NX;   // [B,H,Dk,S]
  unsigned short* ctxb = vt  + NX;   // [B,S,D_MODEL]

  // 1) conversions (+ weight transpose)
  cvt_f32_bf16<<<(int)((NX + 255) / 256), 256, 0, stream>>>(x, xb, (int)NX);
  dim3 tgrid(D_MODEL / 32, D_MODEL / 32);
  cvt_transpose_bf16<<<tgrid, 256, 0, stream>>>(Wq, Wqt, D_MODEL, D_MODEL);
  cvt_transpose_bf16<<<tgrid, 256, 0, stream>>>(Wk, Wkt, D_MODEL, D_MODEL);
  cvt_transpose_bf16<<<tgrid, 256, 0, stream>>>(Wv, Wvt, D_MODEL, D_MODEL);
  cvt_transpose_bf16<<<tgrid, 256, 0, stream>>>(Wo, Wot, D_MODEL, D_MODEL);

  // 2) projections
  dim3 ggrid(TOKENS / MT, D_MODEL / NT);  // 64 x 8
  gemm_bf16<0><<<ggrid, 256, 0, stream>>>(xb, Wqt, bq, qb, TOKENS, D_MODEL, D_MODEL);
  gemm_bf16<0><<<ggrid, 256, 0, stream>>>(xb, Wkt, bk, kb, TOKENS, D_MODEL, D_MODEL);
  gemm_bf16<2><<<ggrid, 256, 0, stream>>>(xb, Wvt, bv, vt, TOKENS, D_MODEL, D_MODEL);

  // 3) flash attention: 64 (b,h) x 128 q-tiles = 8192 waves / 8 per block
  flash_attn<<<(BATCH * N_HEADS * (SEQ / 16)) / 8, 256, 0, stream>>>(qb, kb, vt, ctxb);

  // 4) output projection -> f32 d_out
  gemm_bf16<1><<<ggrid, 256, 0, stream>>>(ctxb, Wot, bo, d_out, TOKENS, D_MODEL, D_MODEL);
}
